// MoEAdapterBranch_86157043958216
// MI455X (gfx1250) — matrix operand order repaired and checked by reading, not compiled, not run on hardware
//
#include <hip/hip_runtime.h>
#include <math.h>

// ---------------------------------------------------------------------------
// MoE adapter branch, MI455X / gfx1250
//   - bf16 WMMA (v_wmma_f32_16x16x32_bf16) for all three GEMMs
//   - Tensor Data Mover (tensor_load_to_lds + s_wait_tensorcnt) for h-tile staging
// ---------------------------------------------------------------------------

typedef __attribute__((ext_vector_type(16))) __bf16 v16bf;
typedef __attribute__((ext_vector_type(8)))  float  v8f;

struct __align__(16) U4 { unsigned int x, y, z, w; };   // trivial 16B POD

// Load 16 contiguous bf16 (32B) -> v16bf  (B-matrix lane: column fixed, K contiguous)
__device__ __forceinline__ v16bf load_b_lane(const __bf16* p) {
    union { v16bf v; U4 q[2]; } u;
    u.q[0] = *(const U4*)(p);
    u.q[1] = *(const U4*)(p + 8);
    return u.v;
}
// A-matrix lane: two 8-bf16 (16B) chunks: K[k0+16h..k0+16h+7] and K[k0+16h+8..k0+16h+15]
__device__ __forceinline__ v16bf load_a_lane(const __bf16* p0, const __bf16* p1) {
    union { v16bf v; U4 q[2]; } u;
    u.q[0] = *(const U4*)(p0);
    u.q[1] = *(const U4*)(p1);
    return u.v;
}
__device__ __forceinline__ v8f wmma_bf16(v16bf a, v16bf b, v8f c) {
    return __builtin_amdgcn_wmma_f32_16x16x32_bf16(false, a, false, b, (short)0, c, false, false);
}
__device__ __forceinline__ v8f vzero8() {
    v8f z = {0.f,0.f,0.f,0.f,0.f,0.f,0.f,0.f};
    return z;
}

// Tensor Data Mover availability (device pass only; host pass parses fallback)
#if defined(__HIP_DEVICE_COMPILE__) && \
    __has_builtin(__builtin_amdgcn_tensor_load_to_lds) && \
    __has_builtin(__builtin_amdgcn_s_wait_tensorcnt)
#define USE_TDM 1
#else
#define USE_TDM 0
#endif

// Problem constants
#define NB   16
#define NT   1025
#define ND   1024
#define NE   8
#define NR   128
#define NC   192
#define NTOK 16384                  // 16 * 1024 tokens (without cls)
#define FULL_SIZE 16793600          // 16*1025*1024

// ---------------------------------------------------------------------------
// init: copy cls tokens, zero the ortho/entropy scalars
__global__ __launch_bounds__(256) void init_kernel(const float* __restrict__ x,
                                                   float* __restrict__ out) {
    int idx = blockIdx.x * 256 + threadIdx.x;     // 16384 = B*D
    int b = idx >> 10, d = idx & 1023;
    size_t o = ((size_t)b * NT << 10) + d;        // (b*1025 + 0)*1024 + d
    out[o] = x[o];
    if (blockIdx.x == 0 && threadIdx.x < 2)
        out[FULL_SIZE + threadIdx.x] = 0.0f;
}

// ---------------------------------------------------------------------------
// conversion passes (fp32 -> bf16)
__global__ __launch_bounds__(256) void convert_h_kernel(const float* __restrict__ x,
                                                        __bf16* __restrict__ hbf) {
    size_t idx = (size_t)blockIdx.x * 256 + threadIdx.x;   // 16*1024*1024
    int b = (int)(idx >> 20);
    int t = (int)((idx >> 10) & 1023);
    int d = (int)(idx & 1023);
    hbf[idx] = (__bf16)x[((size_t)(b * NT + t + 1) << 10) + d];
}
__global__ __launch_bounds__(256) void convert_q_kernel(const float* __restrict__ Q,
                                                        __bf16* __restrict__ Qt) {
    int idx = blockIdx.x * 256 + threadIdx.x;   // 128*1024, transposed (R,D)
    int r = idx >> 10, d = idx & 1023;
    Qt[idx] = (__bf16)Q[d * NR + r];
}
__global__ __launch_bounds__(256) void convert_w_kernel(const float* __restrict__ Wd,
                                                        const float* __restrict__ Wu,
                                                        __bf16* __restrict__ wd,
                                                        __bf16* __restrict__ wu) {
    size_t idx = (size_t)blockIdx.x * 256 + threadIdx.x;   // 8*192*1024
    wd[idx] = (__bf16)Wd[idx];
    wu[idx] = (__bf16)Wu[idx];
}

// ---------------------------------------------------------------------------
// router: z = h@Q via WMMA, then e-normalize -> logits -> softmax -> probs,
// entropy accumulated atomically.  1 block = 16 tokens, 8 waves (256 thr).
__global__ __launch_bounds__(256) void router_kernel(
    const __bf16* __restrict__ hbf, const __bf16* __restrict__ Qt,
    const float* __restrict__ gamma, const float* __restrict__ masks,
    const float* __restrict__ bias, float* __restrict__ probs_out,
    float* __restrict__ entropy_out)
{
    __shared__ float zbuf[16][NR];      // 8 KB
    __shared__ float s_msm[NE][NR];     // 4 KB
    __shared__ float s_gamma[NR];

    const int tid  = threadIdx.x;
    const int lane = tid & 31;
    const int wv   = tid >> 5;          // 0..7 -> r-slice
    const int m    = lane & 15;
    const int kh   = lane >> 4;
    const int tok0 = blockIdx.x * 16;

    if (tid < NR) {                     // softmax(masks, axis=E) per r
        int r = tid;
        float mv[NE]; float mx = -1e30f;
        #pragma unroll
        for (int e = 0; e < NE; ++e) { mv[e] = masks[e * NR + r]; mx = fmaxf(mx, mv[e]); }
        float s = 0.f;
        #pragma unroll
        for (int e = 0; e < NE; ++e) { mv[e] = __expf(mv[e] - mx); s += mv[e]; }
        float inv = 1.f / s;
        #pragma unroll
        for (int e = 0; e < NE; ++e) s_msm[e][r] = mv[e] * inv;
        s_gamma[r] = gamma[r];
    }

    // 16x16 z tile per wave; K = 1024 in 32-steps
    v8f c = vzero8();
    const __bf16* arow = hbf + ((size_t)(tok0 + m) << 10);
    const __bf16* bcol = Qt + ((size_t)(wv * 16 + (lane & 15)) << 10);
    for (int k0 = 0; k0 < ND; k0 += 32) {
        v16bf a = load_a_lane(arow + k0 + kh * 16, arow + k0 + kh * 16 + 8);
        v16bf b = load_b_lane(bcol + k0 + kh * 16);
        c = wmma_bf16(a, b, c);
    }
    #pragma unroll
    for (int v = 0; v < 8; ++v)
        zbuf[v + kh * 8][wv * 16 + (lane & 15)] = c[v];
    __syncthreads();

    if (tid < 16) {                     // per-token tail: tiny
        int row = tid;
        float esum = 1e-6f;
        float acc[NE] = {0.f,0.f,0.f,0.f,0.f,0.f,0.f,0.f};
        for (int r = 0; r < NR; ++r) {
            float zv = zbuf[row][r];
            float ev = zv * zv;
            esum += ev;
            float eg = ev * s_gamma[r];
            #pragma unroll
            for (int e = 0; e < NE; ++e) acc[e] += eg * s_msm[e][r];
        }
        float inv = 1.f / esum;
        float lg[NE]; float mx = -1e30f;
        #pragma unroll
        for (int e = 0; e < NE; ++e) { lg[e] = acc[e] * inv + bias[e]; mx = fmaxf(mx, lg[e]); }
        float s = 0.f;
        #pragma unroll
        for (int e = 0; e < NE; ++e) { lg[e] = __expf(lg[e] - mx); s += lg[e]; }
        float invs = 1.f / s;
        float ent = 0.f;
        #pragma unroll
        for (int e = 0; e < NE; ++e) {
            float p = lg[e] * invs;
            probs_out[(size_t)(tok0 + row) * NE + e] = p;
            ent -= p * __logf(fmaxf(p, 1e-9f));
        }
        atomicAdd(entropy_out, ent * (1.0f / (float)NTOK));
    }
}

// ---------------------------------------------------------------------------
// ortho penalty: lambda * (||Q^T Q - I||_F^2 + ||P^T P - I||_F^2), pure fp32
__global__ __launch_bounds__(256) void ortho_kernel(const float* __restrict__ Q,
                                                    const float* __restrict__ P,
                                                    float* __restrict__ ortho_out)
{
    int idx = blockIdx.x * 256 + threadIdx.x;   // 16384 (i,j) pairs
    int i = idx >> 7, j = idx & 127;
    float sq = 0.f, sp = 0.f;
    for (int d = 0; d < ND; ++d) {
        sq += Q[d * NR + i] * Q[d * NR + j];
        sp += P[d * NR + i] * P[d * NR + j];
    }
    float I = (i == j) ? 1.f : 0.f;
    float v = (sq - I) * (sq - I) + (sp - I) * (sp - I);
    __shared__ float red[256];
    red[threadIdx.x] = v;
    __syncthreads();
    for (int s = 128; s > 0; s >>= 1) {
        if (threadIdx.x < s) red[threadIdx.x] += red[threadIdx.x + s];
        __syncthreads();
    }
    if (threadIdx.x == 0) atomicAdd(ortho_out, 0.001f * red[0]);
}

// ---------------------------------------------------------------------------
// fused expert FFN: per 16-token tile, for each expert:
//   mid = gelu(h @ Wd[e]^T)  (16x192, LDS)  ->  out += probs .* (mid @ Wu[e]^T)
// then y = h + alpha*out written to d_out.  8 waves; wave w owns cols [128w,128w+128).
// h tile (32KB, contiguous) is staged into LDS by the Tensor Data Mover.
__global__ __launch_bounds__(256) void ffn_kernel(
    const __bf16* __restrict__ hbf, const __bf16* __restrict__ Wd,
    const __bf16* __restrict__ Wu, const float* __restrict__ probs,
    const float* __restrict__ x, const float* __restrict__ alpha_p,
    float* __restrict__ out)
{
    __shared__ __align__(16) __bf16 s_h[16][ND];    // 32 KB, reused 16x
    __shared__ __align__(16) __bf16 s_mid[16][NC];  // 6 KB
    __shared__ float s_p[16][NE];                   // 512 B

    const int tid  = threadIdx.x;
    const int lane = tid & 31;
    const int wv   = tid >> 5;
    const int m    = lane & 15;
    const int n16  = lane & 15;
    const int kh   = lane >> 4;
    const int tok0 = blockIdx.x * 16;

#if USE_TDM
    {   // ---- TDM: DMA 16x1024 bf16 (one contiguous 32KB tile) global -> LDS
        typedef __attribute__((ext_vector_type(4))) unsigned int u32x4;
        typedef __attribute__((ext_vector_type(8))) int i32x8;
        typedef __attribute__((ext_vector_type(4))) int i32x4;
        typedef __attribute__((address_space(3))) __bf16 lds_bf16_t;
        if (wv == 0) {   // one wave issues; descriptor is block-uniform -> SGPRs
            unsigned long long gaddr =
                (unsigned long long)(hbf + ((size_t)tok0 << 10));
            unsigned lds_base =
                (unsigned)(unsigned long long)(lds_bf16_t*)&s_h[0][0];
            u32x4 g0;
            g0.x = 1u;                                        // count=1, user desc
            g0.y = lds_base;                                  // lds_addr (bytes)
            g0.z = (unsigned)(gaddr & 0xFFFFFFFFull);         // global_addr[31:0]
            g0.w = (unsigned)((gaddr >> 32) & 0x01FFFFFFull)  // global_addr[56:32]
                 | (2u << 30);                                // type = 2 ("image")
            // 1-D tile: data_size=2B, tensor_dim0 = tile_dim0 = 16384 elems,
            // tensor_dim1 = 1, dim0 stride = 16384, no pad/iterate/multicast.
            i32x8 g1 = { (int)0x00010000,          // data_size=1 (2 bytes)
                         (int)0x40000000,          // tensor_dim0.lo16 << 16 (16384)
                         (int)0x00010000,          // tensor_dim0.hi=0 | tensor_dim1.lo=1
                         (int)0x40000000,          // tensor_dim1.hi=0 | tile_dim0=16384
                         0,                        // tile_dim1=0, tile_dim2=0
                         16384,                    // tensor_dim0_stride.lo32
                         0, 0 };                   // stride hi / dim1_stride
            i32x4 gz4 = {0, 0, 0, 0};
#if __clang_major__ >= 23
            i32x8 gz8 = {0, 0, 0, 0, 0, 0, 0, 0};
            __builtin_amdgcn_tensor_load_to_lds(g0, g1, gz4, gz4, gz8, 0);
#else
            __builtin_amdgcn_tensor_load_to_lds(g0, g1, gz4, gz4, 0);
#endif
            __builtin_amdgcn_s_wait_tensorcnt(0);
        }
    }
#else
    {   // fallback: cooperative 16B copies through VGPRs
        const U4* src = (const U4*)(hbf + ((size_t)tok0 << 10));
        U4* dst = (U4*)&s_h[0][0];
        for (int i = tid; i < 2048; i += 256) dst[i] = src[i];
    }
#endif
    if (tid < 128) ((float*)s_p)[tid] = probs[(size_t)tok0 * NE + tid];
    __syncthreads();   // publishes TDM-written s_h (+ s_p) to all waves

    const float alpha = alpha_p[0];

    v8f acc[8];
    #pragma unroll
    for (int j = 0; j < 8; ++j) acc[j] = vzero8();

    for (int e = 0; e < NE; ++e) {
        // ---- stage 1: mid = gelu(h @ Wd[e]^T); 12 n-tiles over 8 waves
        for (int nt = wv; nt < 12; nt += 8) {
            v8f cm = vzero8();
            const __bf16* arow = &s_h[m][0];
            const __bf16* bcol = Wd + ((size_t)e * NC + nt * 16 + n16) * ND;
            for (int k0 = 0; k0 < ND; k0 += 32) {
                __builtin_prefetch(bcol + k0 + 256, 0, 1);
                v16bf a = load_a_lane(arow + k0 + kh * 16, arow + k0 + kh * 16 + 8);
                v16bf b = load_b_lane(bcol + k0 + kh * 16);
                cm = wmma_bf16(a, b, cm);
            }
            #pragma unroll
            for (int v = 0; v < 8; ++v) {
                float xv = cm[v];
                float g = 0.5f * xv * (1.0f + erff(xv * 0.70710678118654752f));
                s_mid[v + kh * 8][nt * 16 + n16] = (__bf16)g;
            }
        }
        __syncthreads();

        // ---- stage 2: out_tile[j] += probs[:,e] .* (mid @ Wu[e]^T) slice
        #pragma unroll
        for (int j = 0; j < 8; ++j) {
            v8f c2 = vzero8();
            const __bf16* arow = &s_mid[m][0];
            const __bf16* bcol = Wu + ((size_t)e * ND + wv * 128 + j * 16 + n16) * NC;
            #pragma unroll
            for (int k0 = 0; k0 < NC; k0 += 32) {
                v16bf a = load_a_lane(arow + k0 + kh * 16, arow + k0 + kh * 16 + 8);
                v16bf b = load_b_lane(bcol + k0 + kh * 16);
                c2 = wmma_bf16(a, b, c2);
            }
            #pragma unroll
            for (int v = 0; v < 8; ++v)
                acc[j][v] += s_p[v + kh * 8][e] * c2[v];
        }
        __syncthreads();   // protect s_mid before next expert overwrites it
    }

    // ---- residual write: y = h + alpha*out  (skip cls slot)
    #pragma unroll
    for (int j = 0; j < 8; ++j) {
        #pragma unroll
        for (int v = 0; v < 8; ++v) {
            int row   = v + kh * 8;
            int token = tok0 + row;
            int b = token >> 10;
            int t = token & 1023;
            int d = wv * 128 + j * 16 + n16;
            size_t idx = ((size_t)(b * NT + t + 1) << 10) + d;
            out[idx] = x[idx] + alpha * acc[j][v];
        }
    }
}

// ---------------------------------------------------------------------------
extern "C" void kernel_launch(void* const* d_in, const int* in_sizes, int n_in,
                              void* d_out, int out_size, void* d_ws, size_t ws_size,
                              hipStream_t stream) {
    const float* x      = (const float*)d_in[0];
    const float* Q      = (const float*)d_in[1];
    const float* P      = (const float*)d_in[2];
    const float* gamma  = (const float*)d_in[3];
    const float* masks  = (const float*)d_in[4];
    const float* bias   = (const float*)d_in[5];
    const float* W_down = (const float*)d_in[6];
    const float* W_up   = (const float*)d_in[7];
    const float* alpha  = (const float*)d_in[8];
    float* out = (float*)d_out;

    char* ws = (char*)d_ws;
    __bf16* hbf  = (__bf16*)ws; ws += (size_t)NTOK * ND * 2;        // 33.5 MB
    __bf16* Qt   = (__bf16*)ws; ws += (size_t)NR * ND * 2;          // 256 KB
    __bf16* wd   = (__bf16*)ws; ws += (size_t)NE * NC * ND * 2;     // 3 MB
    __bf16* wu   = (__bf16*)ws; ws += (size_t)NE * ND * NC * 2;     // 3 MB
    float*  prb  = (float*)ws;                                      // 512 KB

    init_kernel     <<<64,    256, 0, stream>>>(x, out);
    convert_h_kernel<<<65536, 256, 0, stream>>>(x, hbf);
    convert_q_kernel<<<512,   256, 0, stream>>>(Q, Qt);
    convert_w_kernel<<<6144,  256, 0, stream>>>(W_down, W_up, wd, wu);
    router_kernel   <<<1024,  256, 0, stream>>>(hbf, Qt, gamma, masks, bias,
                                                prb, out + FULL_SIZE + 1);
    ortho_kernel    <<<64,    256, 0, stream>>>(Q, P, out + FULL_SIZE);
    ffn_kernel      <<<1024,  256, 0, stream>>>(hbf, wd, wu, prb, x, alpha, out);
}
